// DeformConvBNAct_4028679323800
// MI455X (gfx1250) — matrix operand order repaired and checked by reading, not compiled, not run on hardware
//
#include <hip/hip_runtime.h>
#include <hip/hip_bf16.h>
#include <cstdint>

typedef __attribute__((ext_vector_type(16))) __bf16 v16bf;
typedef __attribute__((ext_vector_type(8)))  __bf16 v8bf;
typedef __attribute__((ext_vector_type(8)))  float  v8f;
typedef __attribute__((ext_vector_type(4))) unsigned int u32x4;
typedef __attribute__((ext_vector_type(8))) int i32x8;
typedef __attribute__((ext_vector_type(4))) int i32x4;

#define B_     8
#define CIN    128
#define COUT   256
#define HDIM   64
#define WDIM   64
#define HW     4096
#define KK9    9
#define KDIM   1152    // CIN * 9
#define MTOT   32768   // B * H * W
#define KSTEPS 36      // KDIM / 32

// workspace layout (bytes)
#define OFF_OFF   ((size_t)0)          // 8*18*4096*4  = 2359296
#define MOD_OFF   ((size_t)2359296)    // 8*9*4096*4   = 1179648
#define WPACK_OFF ((size_t)3538944)    // 16*36*32*16*2 = 589824
#define ASAMP_OFF ((size_t)4128768)    // 32768*1152*2 = 75497472
#define STATS_OFF ((size_t)79626240)   // 256*2*4      = 2048

__device__ __forceinline__ unsigned short f32_to_bf16(float f) {
  unsigned int u = __float_as_uint(f);
  u += 0x7FFFu + ((u >> 16) & 1u);   // round-to-nearest-even
  return (unsigned short)(u >> 16);
}

// ---------------------------------------------------------------------------
// TDM 2-D tile load: global (bf16 elements, row length tile0, tile1 rows,
// row stride stride0 elements) -> contiguous LDS at lds_off.
// D# bitfields per CDNA5 ISA.
// ---------------------------------------------------------------------------
__device__ __forceinline__ void tdm_load_2d(unsigned lds_off, const void* gptr,
                                            unsigned tile0, unsigned tile1,
                                            unsigned stride0_elems) {
  unsigned long long ga = (unsigned long long)(uintptr_t)gptr;
  u32x4 g0;
  g0.x = 1u;                                        // count=1 (valid), no gather
  g0.y = lds_off;                                   // lds_addr (bytes)
  g0.z = (unsigned)(ga & 0xFFFFFFFFu);              // global_addr[31:0]
  g0.w = (unsigned)((ga >> 32) & 0x01FFFFFFu)       // global_addr[56:32]
       | 0x80000000u;                               // type=2 ("image")
  i32x8 g1;
  g1[0] = (int)(1u << 16);                          // data_size=1 (2B); mask=0
  g1[1] = (int)((tile0 & 0xFFFFu) << 16);           // tensor_dim0[15:0]
  g1[2] = (int)(((tile0 >> 16) & 0xFFFFu)           // tensor_dim0[31:16]
       | ((tile1 & 0xFFFFu) << 16));                // tensor_dim1[15:0]
  g1[3] = (int)(((tile1 >> 16) & 0xFFFFu)           // tensor_dim1[31:16]
       | ((tile0 & 0xFFFFu) << 16));                // tile_dim0
  g1[4] = (int)(tile1 & 0xFFFFu);                   // tile_dim1 (tile_dim2=0)
  g1[5] = (int)stride0_elems;                       // tensor_dim0_stride[31:0]
  g1[6] = 0;                                        // stride0 hi / stride1 lo
  g1[7] = 0;                                        // stride1 hi
  i32x4 z4 = {0, 0, 0, 0};
#if __clang_major__ >= 23
  i32x8 z8 = {0, 0, 0, 0, 0, 0, 0, 0};
  __builtin_amdgcn_tensor_load_to_lds(g0, g1, z4, z4, z8, 0);
#else
  __builtin_amdgcn_tensor_load_to_lds(g0, g1, z4, z4, 0);
#endif
}

// ---------------------------------------------------------------------------
// Kernel 1: offset conv (18 ch) + modulation conv (9 ch, sigmoid applied)
// ---------------------------------------------------------------------------
__global__ void conv_offmod(const float* __restrict__ x,
                            const float* __restrict__ w_off, const float* __restrict__ b_off,
                            const float* __restrict__ w_mod, const float* __restrict__ b_mod,
                            float* __restrict__ off_buf, float* __restrict__ mod_buf) {
  int pix = blockIdx.x * blockDim.x + threadIdx.x;   // 32768 threads
  int b  = pix >> 12;
  int hw = pix & 4095;
  int h = hw >> 6, w = hw & 63;
  float accO[18], accM[9];
#pragma unroll
  for (int o = 0; o < 18; ++o) accO[o] = b_off[o];
#pragma unroll
  for (int o = 0; o < 9;  ++o) accM[o] = b_mod[o];
  const float* xb = x + (size_t)b * CIN * HW;
  for (int c = 0; c < CIN; ++c) {
    float xv[9];
    const float* xc = xb + (size_t)c * HW;
#pragma unroll
    for (int k = 0; k < 9; ++k) {
      int yy = h + (k / 3) - 1;
      int xx = w + (k % 3) - 1;
      bool valid = (yy >= 0) && (yy < HDIM) && (xx >= 0) && (xx < WDIM);
      xv[k] = valid ? xc[yy * WDIM + xx] : 0.f;
    }
#pragma unroll
    for (int o = 0; o < 18; ++o) {
      const float* wp = w_off + ((size_t)o * CIN + c) * 9;
#pragma unroll
      for (int k = 0; k < 9; ++k) accO[o] = fmaf(xv[k], wp[k], accO[o]);
    }
#pragma unroll
    for (int o = 0; o < 9; ++o) {
      const float* wp = w_mod + ((size_t)o * CIN + c) * 9;
#pragma unroll
      for (int k = 0; k < 9; ++k) accM[o] = fmaf(xv[k], wp[k], accM[o]);
    }
  }
#pragma unroll
  for (int o = 0; o < 18; ++o)
    off_buf[((size_t)b * 18 + o) * HW + hw] = accO[o];
#pragma unroll
  for (int o = 0; o < 9; ++o)
    mod_buf[((size_t)b * 9 + o) * HW + hw] = 1.f / (1.f + __expf(-accM[o]));
}

// ---------------------------------------------------------------------------
// Kernel 2: pack w_dc into WMMA B-fragment (32x16 bf16) lane order.
// wpack[((ntile*36+ks)*32+lane)*16 + s], K_local = (L>>4)*16 + 2v + hi,
// K flat = k*128 + c.
// ---------------------------------------------------------------------------
__global__ void pack_w(const float* __restrict__ w_dc,
                       unsigned short* __restrict__ wpack) {
  int t = blockIdx.x * blockDim.x + threadIdx.x;   // 294912 threads
  int s    = t & 15;
  int lane = (t >> 4) & 31;
  int rem  = t >> 9;
  int ks    = rem % KSTEPS;
  int ntile = rem / KSTEPS;                        // 0..15
  int Klocal = (lane >> 4) * 16 + ((s >> 1) * 2) + (s & 1);
  int K = ks * 32 + Klocal;                        // 0..1151
  int N = ntile * 16 + (lane & 15);                // 0..255
  int k = K >> 7;                                  // 0..8
  int c = K & 127;
  float v = w_dc[(size_t)N * KDIM + (size_t)c * 9 + k];
  wpack[t] = f32_to_bf16(v);
}

// ---------------------------------------------------------------------------
// Kernel 3: bilinear sample * modulation -> bf16 im2col Asamp[M][K=k*128+c]
// ---------------------------------------------------------------------------
__global__ void sample_pack(const float* __restrict__ x,
                            const float* __restrict__ off_buf,
                            const float* __restrict__ mod_buf,
                            unsigned short* __restrict__ asamp) {
  int t = blockIdx.x * blockDim.x + threadIdx.x;   // 294912 threads
  int b   = t / (KK9 * HW);
  int rem = t % (KK9 * HW);
  int k   = rem / HW;
  int hw  = rem % HW;
  int h = hw >> 6, w = hw & 63;
  float dy = off_buf[((size_t)b * 18 + 2 * k)     * HW + hw];
  float dx = off_buf[((size_t)b * 18 + 2 * k + 1) * HW + hw];
  float m  = mod_buf[((size_t)b * 9 + k) * HW + hw];
  float py = dy + (float)(h - 1 + k / 3);
  float px = dx + (float)(w - 1 + k % 3);
  float y0f = floorf(py), x0f = floorf(px);
  int y0 = (int)y0f, x0 = (int)x0f;
  float wy1 = py - y0f, wx1 = px - x0f;
  float wy0 = 1.f - wy1, wx0 = 1.f - wx1;
  int y1 = y0 + 1, x1 = x0 + 1;
  bool vy0 = (y0 >= 0) && (y0 < HDIM), vy1 = (y1 >= 0) && (y1 < HDIM);
  bool vx0 = (x0 >= 0) && (x0 < WDIM), vx1 = (x1 >= 0) && (x1 < WDIM);
  int y0c = min(max(y0, 0), HDIM - 1), y1c = min(max(y1, 0), HDIM - 1);
  int x0c = min(max(x0, 0), WDIM - 1), x1c = min(max(x1, 0), WDIM - 1);
  float w00 = wy0 * wx0 * ((vy0 && vx0) ? 1.f : 0.f) * m;
  float w01 = wy0 * wx1 * ((vy0 && vx1) ? 1.f : 0.f) * m;
  float w10 = wy1 * wx0 * ((vy1 && vx0) ? 1.f : 0.f) * m;
  float w11 = wy1 * wx1 * ((vy1 && vx1) ? 1.f : 0.f) * m;
  int i00 = y0c * WDIM + x0c, i01 = y0c * WDIM + x1c;
  int i10 = y1c * WDIM + x0c, i11 = y1c * WDIM + x1c;
  const float* xb = x + (size_t)b * CIN * HW;
  size_t mrow = (size_t)b * HW + hw;
  unsigned short* dst = asamp + mrow * KDIM + (size_t)k * CIN;
  for (int c = 0; c < CIN; ++c) {
    const float* p = xb + (size_t)c * HW;
    float v = w00 * p[i00] + w01 * p[i01] + w10 * p[i10] + w11 * p[i11];
    dst[c] = f32_to_bf16(v);
  }
}

// ---------------------------------------------------------------------------
// Kernel 4: GEMM  out[m,o] = sum_K Asamp[m,K] * W[K,o]   (M=32768,N=256,K=1152)
// bf16 WMMA 16x16x32, f32 accumulate.  8 waves/block; block tile 128M x 64N.
// TDM (tensor_load_to_lds) double-buffers A (128x32 bf16, row stride 1152)
// and B (4x512 bf16 packed fragments, row stride 18432) into LDS; fragments
// are pulled with ds_load_b128; s_wait_tensorcnt + barrier gate the buffers.
// ---------------------------------------------------------------------------
__global__ void gemm_wmma(const unsigned short* __restrict__ asamp,
                          const unsigned short* __restrict__ wpack,
                          float* __restrict__ out) {
  __shared__ unsigned short ldsA[2][128 * 32];   // 2 x 8 KB
  __shared__ unsigned short ldsB[2][4 * 512];    // 2 x 2 KB... (4KB each)

  const int lane   = threadIdx.x & 31;
  const int wave   = threadIdx.x >> 5;
  const int mtile0 = blockIdx.y * 8;           // block covers mtiles mtile0..+7
  const int mtile  = mtile0 + wave;
  const int nt0    = blockIdx.x * 4;           // 0,4,8,12
  const int half   = lane >> 4;
  const int rloc   = (wave * 16 + (lane & 15)) * 32;   // this lane's A row base

  v8f acc0 = {}, acc1 = {}, acc2 = {}, acc3 = {};

  const unsigned short* aBase = asamp + (size_t)(mtile0 * 16) * KDIM;
  const unsigned short* bBase = wpack + (size_t)nt0 * KSTEPS * 512;

  const unsigned offA0 = (unsigned)(uintptr_t)&ldsA[0][0];
  const unsigned offA1 = (unsigned)(uintptr_t)&ldsA[1][0];
  const unsigned offB0 = (unsigned)(uintptr_t)&ldsB[0][0];
  const unsigned offB1 = (unsigned)(uintptr_t)&ldsB[1][0];

  if (wave == 0) {
    tdm_load_2d(offA0, aBase, 32, 128, KDIM);          // A tile, ks=0
    tdm_load_2d(offB0, bBase, 512, 4, KSTEPS * 512);   // B tile, ks=0
  }

  for (int ks = 0; ks < KSTEPS; ++ks) {
    const int buf = ks & 1;
    if (wave == 0) __builtin_amdgcn_s_wait_tensorcnt(0);
    __syncthreads();   // front buffer ready; back buffer free (last read in iter ks-1)

    if (wave == 0 && (ks + 1) < KSTEPS) {
      tdm_load_2d(buf ? offA0 : offA1, aBase + (ks + 1) * 32, 32, 128, KDIM);
      tdm_load_2d(buf ? offB0 : offB1, bBase + (size_t)(ks + 1) * 512, 512, 4,
                  KSTEPS * 512);
    }

    union { v16bf v; v8bf h[2]; } au;
    au.h[0] = *(const v8bf*)&ldsA[buf][rloc + half * 16];
    au.h[1] = *(const v8bf*)&ldsA[buf][rloc + half * 16 + 8];
    const v16bf b0 = *(const v16bf*)&ldsB[buf][0 * 512 + lane * 16];
    const v16bf b1 = *(const v16bf*)&ldsB[buf][1 * 512 + lane * 16];
    const v16bf b2 = *(const v16bf*)&ldsB[buf][2 * 512 + lane * 16];
    const v16bf b3 = *(const v16bf*)&ldsB[buf][3 * 512 + lane * 16];
    acc0 = __builtin_amdgcn_wmma_f32_16x16x32_bf16(false, au.v, false, b0, (short)0, acc0, false, false);
    acc1 = __builtin_amdgcn_wmma_f32_16x16x32_bf16(false, au.v, false, b1, (short)0, acc1, false, false);
    acc2 = __builtin_amdgcn_wmma_f32_16x16x32_bf16(false, au.v, false, b2, (short)0, acc2, false, false);
    acc3 = __builtin_amdgcn_wmma_f32_16x16x32_bf16(false, au.v, false, b3, (short)0, acc3, false, false);
  }

  // C/D layout: VGPR r, lane L -> M = r + 8*half, N = L%16.
  const int mbase = mtile * 16 + 8 * half;
  const int bidx  = mbase >> 12;
  const int hwb   = mbase & 4095;
  v8f accs[4] = {acc0, acc1, acc2, acc3};
#pragma unroll
  for (int n = 0; n < 4; ++n) {
    const int o = (nt0 + n) * 16 + (lane & 15);
    float* p = out + (size_t)bidx * (COUT * HW) + (size_t)o * HW + hwb;
#pragma unroll
    for (int r = 0; r < 8; ++r) p[r] = accs[n][r];
  }
}

// ---------------------------------------------------------------------------
// Kernel 5: per-channel mean / inv-std over (B, H, W)
// ---------------------------------------------------------------------------
__global__ void bn_reduce(const float* __restrict__ out, float* __restrict__ stats) {
  __shared__ float sh1[256];
  __shared__ float sh2[256];
  const int o = blockIdx.x;
  float s = 0.f, s2 = 0.f;
  for (int i = threadIdx.x; i < MTOT; i += 256) {
    int b = i >> 12, hw = i & 4095;
    float v = out[(size_t)b * (COUT * HW) + (size_t)o * HW + hw];
    s += v; s2 += v * v;
  }
  sh1[threadIdx.x] = s; sh2[threadIdx.x] = s2;
  __syncthreads();
  for (int stride = 128; stride > 0; stride >>= 1) {
    if (threadIdx.x < stride) {
      sh1[threadIdx.x] += sh1[threadIdx.x + stride];
      sh2[threadIdx.x] += sh2[threadIdx.x + stride];
    }
    __syncthreads();
  }
  if (threadIdx.x == 0) {
    float mean = sh1[0] * (1.f / MTOT);
    float var  = sh2[0] * (1.f / MTOT) - mean * mean;
    stats[2 * o]     = mean;
    stats[2 * o + 1] = rsqrtf(var + 1e-5f);
  }
}

// ---------------------------------------------------------------------------
// Kernel 6: in-place BN affine + SiLU
// ---------------------------------------------------------------------------
__global__ void bn_act(float* __restrict__ out, const float* __restrict__ stats,
                       const float* __restrict__ gamma, const float* __restrict__ beta) {
  size_t idx = (size_t)blockIdx.x * blockDim.x + threadIdx.x;  // 8388608
  int o = (int)((idx >> 12) & 255);
  float mean = stats[2 * o], invstd = stats[2 * o + 1];
  float v = out[idx];
  float n = (v - mean) * invstd * gamma[o] + beta[o];
  out[idx] = n / (1.f + __expf(-n));   // SiLU
}

// ---------------------------------------------------------------------------
extern "C" void kernel_launch(void* const* d_in, const int* in_sizes, int n_in,
                              void* d_out, int out_size, void* d_ws, size_t ws_size,
                              hipStream_t stream) {
  const float* x     = (const float*)d_in[0];
  const float* w_off = (const float*)d_in[1];
  const float* b_off = (const float*)d_in[2];
  const float* w_mod = (const float*)d_in[3];
  const float* b_mod = (const float*)d_in[4];
  const float* w_dc  = (const float*)d_in[5];
  const float* gamma = (const float*)d_in[6];
  const float* beta  = (const float*)d_in[7];
  float* out = (float*)d_out;
  char*  ws  = (char*)d_ws;

  float*          off_buf = (float*)(ws + OFF_OFF);
  float*          mod_buf = (float*)(ws + MOD_OFF);
  unsigned short* wpack   = (unsigned short*)(ws + WPACK_OFF);
  unsigned short* asamp   = (unsigned short*)(ws + ASAMP_OFF);
  float*          stats   = (float*)(ws + STATS_OFF);

  conv_offmod<<<128, 256, 0, stream>>>(x, w_off, b_off, w_mod, b_mod, off_buf, mod_buf);
  pack_w<<<1152, 256, 0, stream>>>(w_dc, wpack);
  sample_pack<<<1152, 256, 0, stream>>>(x, off_buf, mod_buf, asamp);
  gemm_wmma<<<dim3(4, 256), 256, 0, stream>>>(asamp, wpack, out);
  bn_reduce<<<256, 256, 0, stream>>>(out, stats);
  bn_act<<<32768, 256, 0, stream>>>(out, stats, gamma, beta);
}
